// ColorAwareAttention_46789373723073
// MI455X (gfx1250) — matrix operand order repaired and checked by reading, not compiled, not run on hardware
//
#include <hip/hip_runtime.h>
#include <hip/hip_bf16.h>
#include <math.h>

// ---------------------------------------------------------------------------
// CDNA5 (gfx1250) wave32 WMMA helpers
// ---------------------------------------------------------------------------
typedef __attribute__((ext_vector_type(16))) __bf16 v16bf;
typedef __attribute__((ext_vector_type(8)))  float  v8f;

// Fragment register image: elements 0..7 / 8..15 are each 16 contiguous bytes
// when the source is bf16 and K runs contiguously -> two b128 loads.
union BF16x16 { v16bf v; unsigned short u[16]; uint4 q[2]; };

__device__ __forceinline__ unsigned short f2bf(float f) {
  unsigned int u = __float_as_uint(f);
  u += 0x7FFFu + ((u >> 16) & 1u);   // round-to-nearest-even
  return (unsigned short)(u >> 16);
}

__device__ __forceinline__ v8f vzero8() {
  v8f z;
#pragma unroll
  for (int i = 0; i < 8; ++i) z[i] = 0.0f;
  return z;
}

__device__ __forceinline__ v8f wmma_bf16(v16bf a, v16bf b, v8f c) {
  return __builtin_amdgcn_wmma_f32_16x16x32_bf16(false, a, false, b, (short)0, c,
                                                 false, false);
}

__device__ __forceinline__ uint4 ld4(const unsigned short* p) {
  return *(const uint4*)p;
}

__device__ __forceinline__ float gelu_exact(float v) {
  return 0.5f * v * (1.0f + erff(v * 0.70710678118654752f));
}

// ---------------------------------------------------------------------------
// fp32 -> bf16 elementwise (weights)
// ---------------------------------------------------------------------------
__global__ __launch_bounds__(256) void k_f2bf(
    const float* __restrict__ in, unsigned short* __restrict__ out, int n)
{
  int i = blockIdx.x * 256 + threadIdx.x;
  if (i < n) out[i] = f2bf(in[i]);
}

// ---------------------------------------------------------------------------
// fp32 [B,C,HW] -> bf16 pixel-major [B,HW,Cfull] (channel slice at coff)
// ---------------------------------------------------------------------------
__global__ __launch_bounds__(256) void k_transpose_bf(
    const float* __restrict__ in, unsigned short* __restrict__ out,
    int C, int HW, int Cfull, int coff, int total)
{
  int idx = blockIdx.x * 256 + threadIdx.x;
  if (idx >= total) return;
  const int pix = idx % HW;
  const int c   = (idx / HW) % C;
  const int b   = idx / (HW * C);
  out[((size_t)b * HW + pix) * (size_t)Cfull + coff + c] = f2bf(in[idx]);
}

// ---------------------------------------------------------------------------
// 1x1 conv as GEMM, all-bf16 operands:
//   Y[b,co,n] = sum_ci W[co,ci] * X[b,n,ci] + bias[co]   (Y fp32 NCHW)
// one wave per (16 co x 64 n) tile; every fragment = two b128 loads
// ---------------------------------------------------------------------------
__global__ __launch_bounds__(32) void k_conv1x1_bf(
    const unsigned short* __restrict__ xT,   // [B, N, CI] bf16
    const unsigned short* __restrict__ wbf,  // [CO, CI]  bf16
    const float* __restrict__ bias, float* __restrict__ y,
    int CI, int CO, int N, int ybs)
{
  const int lane = threadIdx.x & 31;
  const int l15 = lane & 15, lh = lane >> 4;
  const int nb  = blockIdx.x * 64;
  const int cob = blockIdx.y * 16;
  const int b   = blockIdx.z;

  v8f acc[4];
#pragma unroll
  for (int t = 0; t < 4; ++t) acc[t] = vzero8();

  const unsigned short* wrow = wbf + (size_t)(cob + l15) * CI;
  const unsigned short* xbase = xT + (size_t)b * N * (size_t)CI;

  for (int kb = 0; kb < CI; kb += 32) {
    BF16x16 a;                                   // A: W[16co x 32ci]
    a.q[0] = ld4(wrow + kb + 16 * lh);
    a.q[1] = ld4(wrow + kb + 16 * lh + 8);
#pragma unroll
    for (int t = 0; t < 4; ++t) {
      const int n = nb + t * 16 + l15;
      const unsigned short* xrow = xbase + (size_t)n * CI + kb + 16 * lh;
      BF16x16 bm;                                // B: X[32ci x 16n]
      bm.q[0] = ld4(xrow);
      bm.q[1] = ld4(xrow + 8);
      acc[t] = wmma_bf16(a.v, bm.v, acc[t]);
    }
  }

  float* yb = y + (size_t)b * (size_t)ybs;
#pragma unroll
  for (int t = 0; t < 4; ++t) {
#pragma unroll
    for (int r = 0; r < 8; ++r) {
      const int co = cob + r + 8 * lh;
      yb[(size_t)co * N + nb + t * 16 + l15] = acc[t][r] + bias[co];
    }
  }
}

// ---------------------------------------------------------------------------
// GroupNorm (+ optional exact GELU), in place over [B,C,HW] fp32.
// Optional fused bf16 output: bfmode 1 = same layout, 2 = pixel-major [B,HW,C]
// ---------------------------------------------------------------------------
__global__ __launch_bounds__(256) void k_groupnorm(
    float* __restrict__ data, const float* __restrict__ sc,
    const float* __restrict__ bi, int C, int HW, int groups, int do_gelu,
    unsigned short* __restrict__ bfout, int bfmode)
{
  const int g = blockIdx.x % groups;
  const int b = blockIdx.x / groups;
  const int cpg = C / groups;
  const size_t base = ((size_t)b * C + (size_t)g * cpg) * (size_t)HW;
  const int count = cpg * HW;
  const int tid = threadIdx.x;

  float s = 0.0f, s2 = 0.0f;
  for (int i = tid; i < count; i += 256) {
    const float v = data[base + i];
    s += v; s2 += v * v;
  }
  __shared__ float sh[256], sh2[256];
  sh[tid] = s; sh2[tid] = s2;
  __syncthreads();
  for (int st = 128; st > 0; st >>= 1) {
    if (tid < st) { sh[tid] += sh[tid + st]; sh2[tid] += sh2[tid + st]; }
    __syncthreads();
  }
  const float inv = 1.0f / (float)count;
  const float mu = sh[0] * inv;
  const float var = fmaxf(sh2[0] * inv - mu * mu, 0.0f);
  const float rstd = rsqrtf(var + 1e-5f);

  for (int i = tid; i < count; i += 256) {
    const int cl = i / HW;
    const int c = g * cpg + cl;
    float v = (data[base + i] - mu) * rstd * sc[c] + bi[c];
    if (do_gelu) v = gelu_exact(v);
    data[base + i] = v;
    if (bfmode == 1) {
      bfout[base + i] = f2bf(v);
    } else if (bfmode == 2) {
      const int pix = i - cl * HW;
      bfout[((size_t)b * HW + pix) * (size_t)C + c] = f2bf(v);
    }
  }
}

// ---------------------------------------------------------------------------
// grouped 3x3 conv, pad=1, stride=1 (8/16 ch per group -> VALU)
// ---------------------------------------------------------------------------
__global__ __launch_bounds__(256) void k_conv3x3_grouped(
    const float* __restrict__ in, const float* __restrict__ w,
    const float* __restrict__ bias, float* __restrict__ out,
    int B, int C, int H, int W, int groups, int total)
{
  int idx = blockIdx.x * 256 + threadIdx.x;
  if (idx >= total) return;
  const int xo = idx % W;
  const int yo = (idx / W) % H;
  const int co = (idx / (W * H)) % C;
  const int b  = idx / (W * H * C);
  const int cpg = C / groups;
  const int g = co / cpg;

  float acc = bias[co];
  for (int ci = 0; ci < cpg; ++ci) {
    const float* ip = in + ((size_t)b * C + g * cpg + ci) * (size_t)(H * W);
    const float* wp = w + ((size_t)co * cpg + ci) * 9;
#pragma unroll
    for (int ky = 0; ky < 3; ++ky) {
      const int yy = yo + ky - 1;
      if (yy < 0 || yy >= H) continue;
#pragma unroll
      for (int kx = 0; kx < 3; ++kx) {
        const int xx = xo + kx - 1;
        if (xx < 0 || xx >= W) continue;
        acc += wp[ky * 3 + kx] * ip[yy * W + xx];
      }
    }
  }
  out[idx] = acc;
}

// ---------------------------------------------------------------------------
// dense 3x3 conv, pad=1, stride=s (region q/k projections, 64->64 ch)
// ---------------------------------------------------------------------------
__global__ __launch_bounds__(256) void k_conv3x3s(
    const float* __restrict__ in, const float* __restrict__ w,
    const float* __restrict__ bias, float* __restrict__ out,
    int B, int CI, int CO, int H, int W, int stride, int total)
{
  int idx = blockIdx.x * 256 + threadIdx.x;
  if (idx >= total) return;
  const int Wo = W / stride, Ho = H / stride;
  const int xo = idx % Wo;
  const int yo = (idx / Wo) % Ho;
  const int co = (idx / (Wo * Ho)) % CO;
  const int b  = idx / (Wo * Ho * CO);
  const int ycent = yo * stride - 1, xcent = xo * stride - 1;

  float acc = bias[co];
  for (int ci = 0; ci < CI; ++ci) {
    const float* ip = in + ((size_t)b * CI + ci) * (size_t)(H * W);
    const float* wp = w + ((size_t)co * CI + ci) * 9;
#pragma unroll
    for (int ky = 0; ky < 3; ++ky) {
      const int yy = ycent + ky;
      if (yy < 0 || yy >= H) continue;
#pragma unroll
      for (int kx = 0; kx < 3; ++kx) {
        const int xx = xcent + kx;
        if (xx < 0 || xx >= W) continue;
        acc += wp[ky * 3 + kx] * ip[yy * W + xx];
      }
    }
  }
  out[idx] = acc;
}

// ---------------------------------------------------------------------------
// bilinear resize (align_corners=False), fp32 NCHW in -> bf16 out
// mode 1: same layout [B,C,Ho,Wo] (batch stride obs)
// mode 2: pixel-major [B,Ho*Wo,Cfull] at channel offset coff (concat slice)
// ---------------------------------------------------------------------------
__global__ __launch_bounds__(256) void k_resize_bf(
    const float* __restrict__ in, unsigned short* __restrict__ out,
    int C, int Hi, int Wi, int Ho, int Wo, int ibs,
    int mode, int obs, int Cfull, int coff, int total)
{
  int idx = blockIdx.x * 256 + threadIdx.x;
  if (idx >= total) return;
  const int xo = idx % Wo;
  const int yo = (idx / Wo) % Ho;
  const int c  = (idx / (Wo * Ho)) % C;
  const int b  = idx / (Wo * Ho * C);

  float fy = ((float)yo + 0.5f) * (float)Hi / (float)Ho - 0.5f;
  float fx = ((float)xo + 0.5f) * (float)Wi / (float)Wo - 0.5f;
  fy = fmaxf(fy, 0.0f); fx = fmaxf(fx, 0.0f);
  int y0 = (int)fy; y0 = min(y0, Hi - 1);
  int x0 = (int)fx; x0 = min(x0, Wi - 1);
  const int y1 = min(y0 + 1, Hi - 1);
  const int x1 = min(x0 + 1, Wi - 1);
  const float wy = fy - (float)y0, wx = fx - (float)x0;

  const float* ip = in + (size_t)b * (size_t)ibs + (size_t)c * (size_t)(Hi * Wi);
  const float v00 = ip[y0 * Wi + x0], v01 = ip[y0 * Wi + x1];
  const float v10 = ip[y1 * Wi + x0], v11 = ip[y1 * Wi + x1];
  const float v = (1.0f - wy) * ((1.0f - wx) * v00 + wx * v01) +
                  wy * ((1.0f - wx) * v10 + wx * v11);
  if (mode == 1)
    out[(size_t)b * (size_t)obs + (size_t)c * (size_t)(Ho * Wo) + yo * Wo + xo] = f2bf(v);
  else
    out[((size_t)b * (Ho * Wo) + yo * Wo + xo) * (size_t)Cfull + coff + c] = f2bf(v);
}

// ---------------------------------------------------------------------------
// attention pass 1: softmax row max / row sum over keys.
// qkT: bf16 [B, N, 64]; one wave per (b, 16-query tile); bf16 WMMA, c=64
// ---------------------------------------------------------------------------
__global__ __launch_bounds__(32) void k_attn_stats(
    const unsigned short* __restrict__ qkT, float* __restrict__ rowmax,
    float* __restrict__ rowsum, const float* __restrict__ scalep, int si, int N)
{
  const int lane = threadIdx.x & 31;
  const int l15 = lane & 15, lh = lane >> 4;
  const int nb = blockIdx.x * 16;
  const int b  = blockIdx.y;
  const float sct = scalep[si] * 0.125f;       // scale_i / sqrt(64)
  const unsigned short* qb = qkT + (size_t)b * N * 64;

  BF16x16 qA[2];                               // A = Q^T [16n x 64c]
  {
    const unsigned short* nrow = qb + (size_t)(nb + l15) * 64;
#pragma unroll
    for (int h = 0; h < 2; ++h) {
      qA[h].q[0] = ld4(nrow + 32 * h + 16 * lh);
      qA[h].q[1] = ld4(nrow + 32 * h + 16 * lh + 8);
    }
  }

  float rm[8], rs[8];
#pragma unroll
  for (int r = 0; r < 8; ++r) { rm[r] = -3.0e38f; rs[r] = 0.0f; }

  for (int mb = 0; mb < N; mb += 16) {
    const unsigned short* mrow = qb + (size_t)(mb + l15) * 64;
    BF16x16 kB[2];                             // B = K [64c x 16m]
#pragma unroll
    for (int h = 0; h < 2; ++h) {
      kB[h].q[0] = ld4(mrow + 32 * h + 16 * lh);
      kB[h].q[1] = ld4(mrow + 32 * h + 16 * lh + 8);
    }
    v8f s = vzero8();
    s = wmma_bf16(qA[0].v, kB[0].v, s);
    s = wmma_bf16(qA[1].v, kB[1].v, s);
#pragma unroll
    for (int r = 0; r < 8; ++r) {
      const float sv = s[r] * sct;
      float cm = sv;                           // chunk max over 16 key lanes
      cm = fmaxf(cm, __shfl_xor(cm, 1, 32));
      cm = fmaxf(cm, __shfl_xor(cm, 2, 32));
      cm = fmaxf(cm, __shfl_xor(cm, 4, 32));
      cm = fmaxf(cm, __shfl_xor(cm, 8, 32));
      const float nm = fmaxf(rm[r], cm);
      float p = __expf(sv - nm);
      p += __shfl_xor(p, 1, 32);
      p += __shfl_xor(p, 2, 32);
      p += __shfl_xor(p, 4, 32);
      p += __shfl_xor(p, 8, 32);
      rs[r] = rs[r] * __expf(rm[r] - nm) + p;
      rm[r] = nm;
    }
  }
  if (l15 == 0) {
#pragma unroll
    for (int r = 0; r < 8; ++r) {
      const int n = nb + r + 8 * lh;
      rowmax[(size_t)b * N + n] = rm[r];
      rowsum[(size_t)b * N + n] = rs[r];
    }
  }
}

// ---------------------------------------------------------------------------
// attention pass 2: O[v,n] = (1/rowsum[n]) * sum_m exp(S-rowmax) * V[v,m]
// qkT bf16 [B,N,64]; V bf16 [B,256,N]; O fp32 [B,256,N].
// 16 waves/WG, 32-query tile; wave w owns 16 V channels. Per 512-key
// super-chunk each wave produces its P chunk into LDS (exact WMMA B-layout,
// 32B-aligned), then all waves consume all chunks with two WMMAs each (one per 16-query tile).
// ---------------------------------------------------------------------------
__global__ __launch_bounds__(512) void k_attn_out(
    const unsigned short* __restrict__ qkT, const unsigned short* __restrict__ vbf,
    float* __restrict__ outp,
    const float* __restrict__ rowmax, const float* __restrict__ rowsum,
    const float* __restrict__ scalep, int si, int N)
{
  __shared__ alignas(32) unsigned short Pbuf[16][2][16][32]; // [chunk][t][n][m]
  __shared__ float rmL[32], rsL[32];

  const int tid = threadIdx.x;
  const int wave = tid >> 5, lane = tid & 31;
  const int l15 = lane & 15, lh = lane >> 4;
  const int nb = blockIdx.x * 32;
  const int b  = blockIdx.y;
  const float sct = scalep[si] * 0.125f;
  const unsigned short* qb = qkT + (size_t)b * N * 64;

  if (tid < 32) {
    rmL[tid] = rowmax[(size_t)b * N + nb + tid];
    rsL[tid] = 1.0f / rowsum[(size_t)b * N + nb + tid];
  }
  __syncthreads();

  BF16x16 qA[2][2];                            // Q^T frags, two 16-query tiles
#pragma unroll
  for (int t = 0; t < 2; ++t) {
    const unsigned short* nrow = qb + (size_t)(nb + t * 16 + l15) * 64;
#pragma unroll
    for (int h = 0; h < 2; ++h) {
      qA[t][h].q[0] = ld4(nrow + 32 * h + 16 * lh);
      qA[t][h].q[1] = ld4(nrow + 32 * h + 16 * lh + 8);
    }
  }

  v8f acc[2];
  acc[0] = vzero8(); acc[1] = vzero8();
  const unsigned short* vrow = vbf + ((size_t)b * 256 + wave * 16 + l15) * (size_t)N;
  const int nchunks = N / 32;

  for (int scb = 0; scb < nchunks; scb += 16) {
    const int myc = scb + wave;
    if (myc < nchunks) {
      const int mb = myc * 32;
      BF16x16 kB[2][2];
#pragma unroll
      for (int mh = 0; mh < 2; ++mh) {
        const unsigned short* mrow = qb + (size_t)(mb + mh * 16 + l15) * 64;
#pragma unroll
        for (int h = 0; h < 2; ++h) {
          kB[mh][h].q[0] = ld4(mrow + 32 * h + 16 * lh);
          kB[mh][h].q[1] = ld4(mrow + 32 * h + 16 * lh + 8);
        }
      }
#pragma unroll
      for (int t = 0; t < 2; ++t)
#pragma unroll
        for (int mh = 0; mh < 2; ++mh) {
          v8f s = vzero8();
          s = wmma_bf16(qA[t][0].v, kB[mh][0].v, s);
          s = wmma_bf16(qA[t][1].v, kB[mh][1].v, s);
#pragma unroll
          for (int r = 0; r < 8; ++r) {
            const int nl = t * 16 + r + 8 * lh;
            const float p = __expf(s[r] * sct - rmL[nl]);
            Pbuf[wave][t][r + 8 * lh][l15 + 16 * mh] = f2bf(p);
          }
        }
    }
    __syncthreads();

    const int nv = min(16, nchunks - scb);
    for (int c = 0; c < nv; ++c) {
      const int mb2 = (scb + c) * 32;
      BF16x16 vA;                              // A = V tile [16v x 32m]
      vA.q[0] = ld4(vrow + mb2 + 16 * lh);
      vA.q[1] = ld4(vrow + mb2 + 16 * lh + 8);
#pragma unroll
      for (int t = 0; t < 2; ++t) {
        const v16bf pb = *(const v16bf*)&Pbuf[c][t][l15][16 * lh]; // 32B LDS load
        acc[t] = wmma_bf16(vA.v, pb, acc[t]);
      }
    }
    __syncthreads();
  }

#pragma unroll
  for (int t = 0; t < 2; ++t) {
    const float rsv = rsL[t * 16 + l15];
    const int n = nb + t * 16 + l15;
#pragma unroll
    for (int r = 0; r < 8; ++r) {
      const int v = wave * 16 + r + 8 * lh;
      outp[((size_t)b * 256 + v) * (size_t)N + n] = acc[t][r] * rsv;
    }
  }
}

// ---------------------------------------------------------------------------
// residual: out = x + gamma * out
// ---------------------------------------------------------------------------
__global__ __launch_bounds__(256) void k_add_skip(
    float* __restrict__ out, const float* __restrict__ x,
    const float* __restrict__ gamma, int total)
{
  int i = blockIdx.x * 256 + threadIdx.x;
  if (i < total) out[i] = x[i] + gamma[0] * out[i];
}

// ---------------------------------------------------------------------------
// host orchestration
// ---------------------------------------------------------------------------
extern "C" void kernel_launch(void* const* d_in, const int* in_sizes, int n_in,
                              void* d_out, int out_size, void* d_ws, size_t ws_size,
                              hipStream_t stream) {
  const int B = 4, C = 256, H = 64, W = 64, HW = H * W;
  auto in = [&](int i) { return (const float*)d_in[i]; };

  const float* x = in(0);
  const float *st_w1 = in(1), *st_b1 = in(2), *st_g1s = in(3), *st_g1b = in(4);
  const float *st_w2 = in(5), *st_b2 = in(6), *st_g2s = in(7), *st_g2b = in(8);
  // color proj (9..16) is dead code in the reference
  const float *v_w1 = in(29), *v_b1 = in(30), *v_g1s = in(31), *v_g1b = in(32);
  const float *v_w2 = in(33), *v_b2 = in(34), *v_g2s = in(35), *v_g2b = in(36);
  const float *o_w1 = in(37), *o_b1 = in(38), *o_gs = in(39), *o_gb = in(40);
  const float *o_w2 = in(41), *o_b2 = in(42);
  const float *scalep = in(43), *gammap = in(44);

  // workspace carve-up (bytes -> typed, all chunks 16B aligned)
  char* ws = (char*)d_ws;
  size_t off = 0;
  auto allocf = [&](size_t n) { float* p = (float*)(ws + off); off += n * 4; return p; };
  auto allocb = [&](size_t n) { unsigned short* p = (unsigned short*)(ws + off);
                                off += ((n * 2 + 15) & ~(size_t)15); return p; };
  float* rowmax = allocf((size_t)B * HW);
  float* rowsum = allocf((size_t)B * HW);
  float* t0    = allocf((size_t)B * 64 * HW);    // struct hidden (fp32)
  float* t1    = allocf((size_t)B * 64 * HW);    // struct (fp32)
  float* tvh   = allocf((size_t)B * 256 * HW);   // value hidden (fp32)
  float* tval  = allocf((size_t)B * 256 * HW);   // value (fp32)
  float* tqk   = allocf((size_t)B * 64 * HW);    // region q/k (fp32)
  float* to1   = allocf((size_t)B * 256 * HW);   // out-proj hidden (fp32)
  float* tattn = allocf((size_t)B * 256 * HW);   // attention out (fp32)
  unsigned short* xT    = allocb((size_t)B * HW * 256);   // x, pixel-major bf16
  unsigned short* qkT   = allocb((size_t)B * HW * 64);    // qk, pixel-major bf16
  unsigned short* valbf = allocb((size_t)B * 256 * HW);   // value, ch-major bf16
  unsigned short* tvsbf = allocb((size_t)B * 256 * 1024); // downsized value bf16
  unsigned short* comb  = allocb((size_t)B * HW * 768);   // concat, pixel-major
  unsigned short* to1T  = allocb((size_t)B * HW * 256);   // hidden, pixel-major
  unsigned short* w1bf  = allocb(64 * 256);
  unsigned short* w2bf  = allocb(256 * 256);
  unsigned short* w3bf  = allocb(256 * 768);
  unsigned short* w4bf  = allocb(256 * 256);

  int tot;

  // ---- weight + input conversions ----
  k_f2bf<<<(64 * 256 + 255) / 256, 256, 0, stream>>>(st_w1, w1bf, 64 * 256);
  k_f2bf<<<(256 * 256 + 255) / 256, 256, 0, stream>>>(v_w1, w2bf, 256 * 256);
  k_f2bf<<<(256 * 768 + 255) / 256, 256, 0, stream>>>(o_w1, w3bf, 256 * 768);
  k_f2bf<<<(256 * 256 + 255) / 256, 256, 0, stream>>>(o_w2, w4bf, 256 * 256);
  tot = B * 256 * HW;
  k_transpose_bf<<<(tot + 255) / 256, 256, 0, stream>>>(x, xT, 256, HW, 256, 0, tot);

  // ---- structure projection ----
  k_conv1x1_bf<<<dim3(HW / 64, 4, B), 32, 0, stream>>>(xT, w1bf, st_b1, t0,
      256, 64, HW, 64 * HW);
  k_groupnorm<<<B * 8, 256, 0, stream>>>(t0, st_g1s, st_g1b, 64, HW, 8, 1,
      (unsigned short*)nullptr, 0);
  tot = B * 64 * HW;
  k_conv3x3_grouped<<<(tot + 255) / 256, 256, 0, stream>>>(t0, st_w2, st_b2, t1,
      B, 64, H, W, 8, tot);
  k_groupnorm<<<B * 8, 256, 0, stream>>>(t1, st_g2s, st_g2b, 64, HW, 8, 1,
      (unsigned short*)nullptr, 0);

  // ---- value projection ----
  k_conv1x1_bf<<<dim3(HW / 64, 16, B), 32, 0, stream>>>(xT, w2bf, v_b1, tvh,
      256, 256, HW, 256 * HW);
  k_groupnorm<<<B * 16, 256, 0, stream>>>(tvh, v_g1s, v_g1b, 256, HW, 16, 0,
      (unsigned short*)nullptr, 0);
  tot = B * 256 * HW;
  k_conv3x3_grouped<<<(tot + 255) / 256, 256, 0, stream>>>(tvh, v_w2, v_b2, tval,
      B, 256, H, W, 16, tot);
  k_groupnorm<<<B * 16, 256, 0, stream>>>(tval, v_g2s, v_g2b, 256, HW, 16, 1,
      valbf, 1);

  // ---- region attention branches (s = 1, 2, 4) ----
  for (int i = 0; i < 3; ++i) {
    const int s = 1 << i;
    const int h = 64 / s, Ns = h * h;
    const float *rw = in(17 + 4 * i), *rb = in(18 + 4 * i);
    const float *rgs = in(19 + 4 * i), *rgb = in(20 + 4 * i);

    tot = B * 64 * Ns;
    k_conv3x3s<<<(tot + 255) / 256, 256, 0, stream>>>(t1, rw, rb, tqk,
        B, 64, 64, H, W, s, tot);
    // GN + GELU, fused bf16 pixel-major output for the WMMA attention
    k_groupnorm<<<B * 8, 256, 0, stream>>>(tqk, rgs, rgb, 64, Ns, 8, 1, qkT, 2);

    const unsigned short* vsp = valbf;
    if (s > 1) {
      tot = B * 256 * Ns;
      k_resize_bf<<<(tot + 255) / 256, 256, 0, stream>>>(tval, tvsbf,
          256, 64, 64, h, h, 256 * HW, /*mode=*/1, 256 * Ns, 0, 0, tot);
      vsp = tvsbf;
    }

    k_attn_stats<<<dim3(Ns / 16, B), 32, 0, stream>>>(qkT, rowmax, rowsum,
        scalep, i, Ns);
    k_attn_out<<<dim3(Ns / 32, B), 512, 0, stream>>>(qkT, vsp, tattn,
        rowmax, rowsum, scalep, i, Ns);

    if (s == 1) {
      tot = B * 256 * HW;
      k_transpose_bf<<<(tot + 255) / 256, 256, 0, stream>>>(tattn, comb,
          256, HW, 768, 0, tot);
    } else {
      tot = B * 256 * HW;
      k_resize_bf<<<(tot + 255) / 256, 256, 0, stream>>>(tattn, comb,
          256, h, h, 64, 64, 256 * Ns, /*mode=*/2, 0, 768, 256 * i, tot);
    }
  }

  // ---- output projection + residual ----
  k_conv1x1_bf<<<dim3(HW / 64, 16, B), 32, 0, stream>>>(comb, w3bf, o_b1, to1,
      768, 256, HW, 256 * HW);
  k_groupnorm<<<B * 16, 256, 0, stream>>>(to1, o_gs, o_gb, 256, HW, 16, 1,
      to1T, 2);
  k_conv1x1_bf<<<dim3(HW / 64, 16, B), 32, 0, stream>>>(to1T, w4bf, o_b2,
      (float*)d_out, 256, 256, HW, 256 * HW);
  tot = B * 256 * HW;
  k_add_skip<<<(tot + 255) / 256, 256, 0, stream>>>((float*)d_out, x, gammap, tot);
}
